// GCNSkeletonTokenizer_10p_1125281431600
// MI455X (gfx1250) — matrix operand order repaired and checked by reading, not compiled, not run on hardware
//
#include <hip/hip_runtime.h>
#include <hip/hip_bf16.h>

// Problem constants
#define G_ 10
#define B_ 32768
#define K_ 1024
#define D_ 256
#define LSTRIDE 132              // u32 per staged code row in LDS: 128 data + 4 pad
#define PLANE   (32 * LSTRIDE)   // u32 per LDS plane (32 code rows)
#define STAGE_U32 (2 * 32 * 128) // u32 per packed stage in ws: hi plane + lo plane, unpadded

typedef __attribute__((ext_vector_type(16))) __bf16 v16bf;
typedef __attribute__((ext_vector_type(8)))  float  v8f;
typedef __attribute__((ext_vector_type(4)))  int    v4i;

__constant__ float c_commit[G_] = {0.5f,0.5f,0.4f,0.4f,0.4f,0.4f,0.8f,0.8f,0.8f,0.8f};

#if defined(__has_builtin)
#if __has_builtin(__builtin_amdgcn_global_load_async_to_lds_b128)
#define HAVE_ASYNC_LDS 1
#endif
#endif

#if HAVE_ASYNC_LDS
typedef __attribute__((address_space(1))) v4i* as1_v4i_ptr;
typedef __attribute__((address_space(3))) v4i* as3_v4i_ptr;
#define WAIT_STAGE() asm volatile("s_wait_asynccnt 0" ::: "memory")
#else
#define WAIT_STAGE() ((void)0)
#endif

// ---------- kernel 0: one-time codebook prep ----------
// Packs codebook into bf16 hi/lo planes laid out per 32-code stage (unpadded rows of
// 128 dwords), and computes per-code squared norms. One wave per code row.
__global__ void __launch_bounds__(256)
vq_prep(const float* __restrict__ codebook, unsigned* __restrict__ pack,
        float* __restrict__ norms) {
  const int t = threadIdx.x;
  const int wave = t >> 5, lane = t & 31;
  const int r = blockIdx.x * 8 + wave;            // code id 0..G_*K_-1
  const int g = r >> 10, k = r & (K_ - 1);
  const int s = k >> 5, rl = k & 31;

  const float* cr = codebook + (size_t)r * D_ + lane * 8;
  float x[8];
  *(float4*)&x[0] = *(const float4*)cr;
  *(float4*)&x[4] = *(const float4*)(cr + 4);

  uint4 hi, lo;
  unsigned* hp = &hi.x;
  unsigned* lp = &lo.x;
  float ss = 0.0f;
  #pragma unroll
  for (int j = 0; j < 4; ++j) {
    float a = x[2*j], b = x[2*j+1];
    ss += a*a + b*b;
    __bf16 ah = (__bf16)a, bh = (__bf16)b;
    __bf16 al = (__bf16)(a - (float)ah), bl = (__bf16)(b - (float)bh);
    unsigned short uah, ubh, ual, ubl;
    __builtin_memcpy(&uah, &ah, 2); __builtin_memcpy(&ubh, &bh, 2);
    __builtin_memcpy(&ual, &al, 2); __builtin_memcpy(&ubl, &bl, 2);
    hp[j] = (unsigned)uah | ((unsigned)ubh << 16);
    lp[j] = (unsigned)ual | ((unsigned)ubl << 16);
  }
  unsigned* dst = pack + (size_t)(g * 32 + s) * STAGE_U32 + rl * 128 + lane * 4;
  *(uint4*)dst          = hi;
  *(uint4*)(dst + 4096) = lo;   // lo plane = second half of stage

  #pragma unroll
  for (int off = 16; off > 0; off >>= 1) ss += __shfl_xor(ss, off, 32);
  if (lane == 0) norms[r] = ss;
}

// Issue one stage copy (32 codes, hi+lo planes) from packed ws into padded LDS planes.
// 256 threads x 8 b128 chunks = 2048 x 16B = 32 KB. Async DMA: no VGPR round-trip.
__device__ __forceinline__ void stage_issue(const unsigned* __restrict__ gsrc,
                                            unsigned* sHi, unsigned* sLo, int t) {
  #pragma unroll
  for (int j = 0; j < 8; ++j) {
    const int c   = j * 256 + t;       // 0..2047
    const int row = c >> 5;            // 0..63 (0..31: hi plane, 32..63: lo plane)
    const int cq  = (c & 31) * 4;      // dword offset within row
    const unsigned* ga = gsrc + row * 128 + cq;
    unsigned* la = (row < 32) ? (sHi + row * LSTRIDE + cq)
                              : (sLo + (row - 32) * LSTRIDE + cq);
#if HAVE_ASYNC_LDS
    __builtin_amdgcn_global_load_async_to_lds_b128((as1_v4i_ptr)ga, (as3_v4i_ptr)la, 0, 0);
#else
    *(uint4*)la = *(const uint4*)ga;
#endif
  }
}

__device__ __forceinline__ v16bf ld_frag(const unsigned* p) {
  union { v16bf v; uint4 q[2]; } u;
  u.q[0] = *(const uint4*)p;
  u.q[1] = *(const uint4*)(p + 4);
  return u.v;
}

// ---------- kernel 1: WMMA distance + argmin ----------
// grid: (B_/128, G_), block 256 (8 waves). Wave w owns rows tile*128 + 16w .. +15.
__global__ void __launch_bounds__(256)
vq_argmin(const float* __restrict__ feats, const unsigned* __restrict__ pack,
          const float* __restrict__ norms, float* __restrict__ idx_out) {
  const int g    = blockIdx.y;
  const int tile = blockIdx.x;
  const int t    = threadIdx.x;
  const int wave = t >> 5, lane = t & 31;
  const int half = lane >> 4, mlane = lane & 15;
  const int rowBase = tile * 128 + wave * 16;
  const int row = rowBase + mlane;

  const float* frow = feats + ((size_t)g * B_ + row) * D_;

  // A fragments (feats rows as bf16 hi/lo), resident in VGPRs for the whole K loop
  v16bf Ahi[8], Alo[8];
  #pragma unroll
  for (int c = 0; c < 8; ++c) {
    const int d0 = c * 32 + half * 16;      // elements 0..7
    const int d1 = c * 32 + half * 16 + 8;  // elements 8..15
    float xs[16];
    *(float4*)&xs[0]  = *(const float4*)(frow + d0);
    *(float4*)&xs[4]  = *(const float4*)(frow + d0 + 4);
    *(float4*)&xs[8]  = *(const float4*)(frow + d1);
    *(float4*)&xs[12] = *(const float4*)(frow + d1 + 4);
    #pragma unroll
    for (int e = 0; e < 16; ++e) {
      __bf16 hb = (__bf16)xs[e];
      Ahi[c][e] = hb;
      Alo[c][e] = (__bf16)(xs[e] - (float)hb);
    }
  }

  float minv[8];
  int   mini[8];
  #pragma unroll
  for (int v = 0; v < 8; ++v) { minv[v] = 3.4e38f; mini[v] = 0; }

  // Dynamic LDS: two buffers x (hi plane + lo plane), padded rows
  extern __shared__ unsigned smem[];
  unsigned* h0 = smem;
  unsigned* l0 = smem + PLANE;
  unsigned* h1 = smem + 2 * PLANE;
  unsigned* l1 = smem + 3 * PLANE;

  const unsigned* packg = pack + (size_t)g * 32 * STAGE_U32;

  // prologue: stage 0
  stage_issue(packg, h0, l0, t);
  WAIT_STAGE();
  __syncthreads();

  int p = 0;
  for (int s = 0; s < K_ / 32; ++s) {
    unsigned* curH = p ? h1 : h0;
    unsigned* curL = p ? l1 : l0;
    if (s + 1 < K_ / 32)   // prefetch next stage into the other buffer (async DMA)
      stage_issue(packg + (size_t)(s + 1) * STAGE_U32, p ? h0 : h1, p ? l0 : l1, t);

    float nrm01[2];
    nrm01[0] = norms[g * K_ + s * 32 + mlane];
    nrm01[1] = norms[g * K_ + s * 32 + 16 + mlane];

    #pragma unroll
    for (int kt = 0; kt < 2; ++kt) {
      const int rowL = kt * 16 + mlane;
      const unsigned* bh = curH + rowL * LSTRIDE;
      const unsigned* bl = curL + rowL * LSTRIDE;

      // software-pipelined B fragments: load chunk c+1 before chunk c's WMMAs
      v8f acc_hh = {}, acc_x = {};
      v16bf fh[2], fl[2];
      fh[0] = ld_frag(bh + half * 8);
      fl[0] = ld_frag(bl + half * 8);
      #pragma unroll
      for (int c = 0; c < 8; ++c) {
        const int cur = c & 1, nxt = cur ^ 1;
        if (c < 7) {
          const int du = (c + 1) * 16 + half * 8;  // dword offset: d = 32(c+1) + 16*half
          fh[nxt] = ld_frag(bh + du);
          fl[nxt] = ld_frag(bl + du);
        }
        acc_hh = __builtin_amdgcn_wmma_f32_16x16x32_bf16(false, Ahi[c], false, fh[cur],
                                                         (short)0, acc_hh, false, false);
        acc_x  = __builtin_amdgcn_wmma_f32_16x16x32_bf16(false, Ahi[c], false, fl[cur],
                                                         (short)0, acc_x, false, false);
        acc_x  = __builtin_amdgcn_wmma_f32_16x16x32_bf16(false, Alo[c], false, fh[cur],
                                                         (short)0, acc_x, false, false);
      }
      const float nrm = nrm01[kt];
      const int kidx = s * 32 + kt * 16 + mlane;
      #pragma unroll
      for (int v = 0; v < 8; ++v) {
        float d2 = nrm - 2.0f * (acc_hh[v] + acc_x[v]);
        if (d2 < minv[v]) { minv[v] = d2; mini[v] = kidx; }
      }
    }
    WAIT_STAGE();          // own async copies done (after compute -> overlapped)
    __syncthreads();       // all waves' copies visible; prev buffer free
    p ^= 1;
  }

  // argmin across the 16 lanes of each half-wave (columns of the C tile)
  #pragma unroll
  for (int off = 8; off > 0; off >>= 1) {
    #pragma unroll
    for (int v = 0; v < 8; ++v) {
      float ov = __shfl_xor(minv[v], off, 16);
      int   oi = __shfl_xor(mini[v], off, 16);
      if (ov < minv[v] || (ov == minv[v] && oi < mini[v])) { minv[v] = ov; mini[v] = oi; }
    }
  }
  if (mlane == 0) {
    const int rb = rowBase + half * 8;   // lane0 -> rows 0..7, lane16 -> rows 8..15
    #pragma unroll
    for (int v = 0; v < 8; ++v)
      idx_out[(size_t)g * B_ + rb + v] = (float)mini[v];
  }
}

// ---------- kernel 2: gather nearest codes, quant_st, loss partials ----------
__global__ void __launch_bounds__(256)
vq_gather(const float* __restrict__ feats, const float* __restrict__ codebook,
          const float* __restrict__ idxf, float* __restrict__ quant_out,
          float* __restrict__ partials) {
  const int t = threadIdx.x;
  const int rl = t >> 5, lane = t & 31;
  const int gb = blockIdx.x * 8 + rl;           // 0 .. G_*B_-1
  const int g = gb >> 15;                       // B_ = 2^15
  const int idx = (int)idxf[gb];

  const float* fr = feats + (size_t)gb * D_;
  const float* cr = codebook + ((size_t)g * K_ + idx) * D_;
  float*       qo = quant_out + (size_t)gb * D_;

  float acc = 0.0f;
  #pragma unroll
  for (int j = 0; j < 2; ++j) {
    const int d = lane * 8 + j * 4;
    float4 c4 = *(const float4*)(cr + d);
    float4 f4 = *(const float4*)(fr + d);
    float4 o4;                                   // straight-through: f + (c - f)
    o4.x = f4.x + (c4.x - f4.x);
    o4.y = f4.y + (c4.y - f4.y);
    o4.z = f4.z + (c4.z - f4.z);
    o4.w = f4.w + (c4.w - f4.w);
    *(float4*)(qo + d) = o4;
    float dx = c4.x - f4.x, dy = c4.y - f4.y, dz = c4.z - f4.z, dw = c4.w - f4.w;
    acc += dx*dx + dy*dy + dz*dz + dw*dw;
  }
  acc *= (1.0f + c_commit[g]) * (1.0f / ((float)B_ * (float)D_));

  __shared__ float red[256];
  red[t] = acc;
  __syncthreads();
  #pragma unroll
  for (int s = 128; s > 0; s >>= 1) {
    if (t < s) red[t] += red[t + s];
    __syncthreads();
  }
  if (t == 0) partials[blockIdx.x] = red[0];
}

// ---------- kernel 3: deterministic final loss reduction ----------
__global__ void vq_loss(const float* __restrict__ partials, int n, float* __restrict__ loss) {
  const int t = threadIdx.x;
  float s = 0.0f;
  for (int i = t; i < n; i += 256) s += partials[i];
  __shared__ float red[256];
  red[t] = s;
  __syncthreads();
  #pragma unroll
  for (int k = 128; k > 0; k >>= 1) {
    if (t < k) red[t] += red[t + k];
    __syncthreads();
  }
  if (t == 0) *loss = red[0];
}

extern "C" void kernel_launch(void* const* d_in, const int* in_sizes, int n_in,
                              void* d_out, int out_size, void* d_ws, size_t ws_size,
                              hipStream_t stream) {
  const float* feats    = (const float*)d_in[0];   // (G,B,D) f32
  const float* codebook = (const float*)d_in[1];   // (G,K,D) f32

  float* out = (float*)d_out;
  const size_t Q = (size_t)G_ * B_ * D_;
  float* quant = out;           // (G,B,D)
  float* loss  = out + Q;       // scalar
  float* idxf  = out + Q + 1;   // (G,B) as float

  // workspace: packed bf16 planes (10 MB) | norms | partials
  unsigned* pack   = (unsigned*)d_ws;
  float* norms     = (float*)(pack + (size_t)G_ * 32 * STAGE_U32);
  float* partials  = norms + G_ * K_;

  vq_prep<<<G_ * K_ / 8, 256, 0, stream>>>(codebook, pack, norms);

  dim3 grid(B_ / 128, G_);
  const size_t smem_bytes = (size_t)4 * PLANE * sizeof(unsigned);  // 67584 B
  vq_argmin<<<grid, 256, smem_bytes, stream>>>(feats, pack, norms, idxf);

  vq_gather<<<G_ * B_ / 8, 256, 0, stream>>>(feats, codebook, idxf, quant, partials);

  vq_loss<<<1, 256, 0, stream>>>(partials, G_ * B_ / 8, loss);
}
